// PointTransformerLast_49804440764997
// MI455X (gfx1250) — matrix operand order repaired and checked by reading, not compiled, not run on hardware
//
#include <hip/hip_runtime.h>

#define BSZ 4
#define CSZ 128
#define NSZ 4096
#define NEGV (-9.0e15f)

typedef __attribute__((ext_vector_type(16))) _Float16 v16h;
typedef __attribute__((ext_vector_type(8)))  _Float16 v8h;
typedef __attribute__((ext_vector_type(8)))  float    v8f;

__device__ __forceinline__ v16h pack16(v8h lo, v8h hi) {
    v16h r;
#pragma unroll
    for (int j = 0; j < 8; ++j) { r[j] = lo[j]; r[8 + j] = hi[j]; }
    return r;
}

// ---------------------------------------------------------------------------
// 128x128 projection: Y[b,o,n] = sum_c W[o,c] * X[b,c,n] (+ bias[o])
// Each thread produces 16 output channels for one n, so every X element is
// loaded once per 16-channel block (16x less traffic).  Weight tile in LDS
// (uniform-address broadcast reads).  grid = b(2) | nblk(4) | oblk(3)
// ---------------------------------------------------------------------------
__global__ void proj128_kernel(const float* __restrict__ W,
                               const float* __restrict__ X,
                               const float* __restrict__ bias,
                               float* __restrict__ Y) {
    __shared__ float wtile[16 * CSZ];
    int id = blockIdx.x;
    const int oblk = id & 7;  id >>= 3;
    const int nblk = id & 15; id >>= 4;
    const int b = id;
    const int o0 = oblk * 16;
    const int n = nblk * 256 + threadIdx.x;

    for (int i = threadIdx.x; i < 16 * CSZ; i += 256)
        wtile[i] = W[(size_t)(o0 + (i >> 7)) * CSZ + (i & 127)];
    __syncthreads();

    const float* xb = X + (size_t)b * CSZ * NSZ + n;
    float acc[16];
#pragma unroll
    for (int k = 0; k < 16; ++k) acc[k] = bias ? bias[o0 + k] : 0.0f;
    for (int c = 0; c < CSZ; ++c) {
        const float xv = xb[(size_t)c * NSZ];
#pragma unroll
        for (int k = 0; k < 16; ++k) acc[k] = fmaf(wtile[(k << 7) + c], xv, acc[k]);
    }
    float* yb = Y + (size_t)b * CSZ * NSZ + n;
#pragma unroll
    for (int k = 0; k < 16; ++k) yb[(size_t)(o0 + k) * NSZ] = acc[k];
}

// Same, but f16 output (xv = wv@h + bv : A-operand of x_r GEMM)
__global__ void proj128_f16_kernel(const float* __restrict__ W,
                                   const float* __restrict__ X,
                                   const float* __restrict__ bias,
                                   _Float16* __restrict__ Y) {
    __shared__ float wtile[16 * CSZ];
    int id = blockIdx.x;
    const int oblk = id & 7;  id >>= 3;
    const int nblk = id & 15; id >>= 4;
    const int b = id;
    const int o0 = oblk * 16;
    const int n = nblk * 256 + threadIdx.x;

    for (int i = threadIdx.x; i < 16 * CSZ; i += 256)
        wtile[i] = W[(size_t)(o0 + (i >> 7)) * CSZ + (i & 127)];
    __syncthreads();

    const float* xb = X + (size_t)b * CSZ * NSZ + n;
    float acc[16];
#pragma unroll
    for (int k = 0; k < 16; ++k) acc[k] = bias[o0 + k];
    for (int c = 0; c < CSZ; ++c) {
        const float xv = xb[(size_t)c * NSZ];
#pragma unroll
        for (int k = 0; k < 16; ++k) acc[k] = fmaf(wtile[(k << 7) + c], xv, acc[k]);
    }
    _Float16* yb = Y + (size_t)b * CSZ * NSZ + n;
#pragma unroll
    for (int k = 0; k < 16; ++k) yb[(size_t)(o0 + k) * NSZ] = (_Float16)acc[k];
}

// q projection, O=32, written TRANSPOSED: qT[b, n, o] (f16, o contiguous).
// 16 channels per thread; vector (2x v8h = 32B) store per thread.
// grid = b(2) | nblk(4) | oblk(1)
__global__ void qproj_kernel(const float* __restrict__ Wqk,
                             const float* __restrict__ X,
                             _Float16* __restrict__ qT) {
    __shared__ float wtile[16 * CSZ];
    int id = blockIdx.x;
    const int oblk = id & 1;  id >>= 1;
    const int nblk = id & 15; id >>= 4;
    const int b = id;
    const int o0 = oblk * 16;
    const int n = nblk * 256 + threadIdx.x;

    for (int i = threadIdx.x; i < 16 * CSZ; i += 256)
        wtile[i] = Wqk[(size_t)(o0 + (i >> 7)) * CSZ + (i & 127)];
    __syncthreads();

    const float* xb = X + (size_t)b * CSZ * NSZ + n;
    float acc[16];
#pragma unroll
    for (int k = 0; k < 16; ++k) acc[k] = 0.0f;
    for (int c = 0; c < CSZ; ++c) {
        const float xv = xb[(size_t)c * NSZ];
#pragma unroll
        for (int k = 0; k < 16; ++k) acc[k] = fmaf(wtile[(k << 7) + c], xv, acc[k]);
    }
    v8h lo, hi;
#pragma unroll
    for (int k = 0; k < 8; ++k) { lo[k] = (_Float16)acc[k]; hi[k] = (_Float16)acc[8 + k]; }
    v8h* dst = (v8h*)(qT + ((size_t)b * NSZ + n) * 32 + o0);
    dst[0] = lo;
    dst[1] = hi;
}

// ---------------------------------------------------------------------------
// Per-channel training-mode BN stats over (B, N): mean / biased var
// ---------------------------------------------------------------------------
__global__ void bnstats_kernel(const float* __restrict__ Y,
                               float* __restrict__ mean,
                               float* __restrict__ var) {
    const int o = blockIdx.x;
    __shared__ float ssum[256];
    __shared__ float ssq[256];
    float s = 0.0f, q = 0.0f;
    for (int b = 0; b < BSZ; ++b) {
        const float* yb = Y + (size_t)b * CSZ * NSZ + (size_t)o * NSZ;
        for (int n = threadIdx.x; n < NSZ; n += 256) {
            float v = yb[n];
            s += v;
            q = fmaf(v, v, q);
        }
    }
    ssum[threadIdx.x] = s; ssq[threadIdx.x] = q;
    __syncthreads();
    for (int st = 128; st > 0; st >>= 1) {
        if (threadIdx.x < st) {
            ssum[threadIdx.x] += ssum[threadIdx.x + st];
            ssq[threadIdx.x]  += ssq[threadIdx.x + st];
        }
        __syncthreads();
    }
    if (threadIdx.x == 0) {
        const float inv = 1.0f / (float)(BSZ * NSZ);
        float m = ssum[0] * inv;
        mean[o] = m;
        var[o] = ssq[0] * inv - m * m;
    }
}

// H = relu(BN(Y))
__global__ void bnrelu_kernel(const float* __restrict__ Y,
                              const float* __restrict__ mean,
                              const float* __restrict__ var,
                              const float* __restrict__ g,
                              const float* __restrict__ bb,
                              float* __restrict__ H) {
    const int idx = blockIdx.x * 256 + threadIdx.x;
    const int c = (idx >> 12) & 127;
    float xh = (Y[idx] - mean[c]) * rsqrtf(var[c] + 1e-5f);
    H[idx] = fmaxf(fmaf(xh, g[c], bb[c]), 0.0f);
}

// h += relu(BN(T)); write h and the layer's output slice
__global__ void residual_kernel(const float* __restrict__ T,
                                const float* __restrict__ mean,
                                const float* __restrict__ var,
                                const float* __restrict__ g,
                                const float* __restrict__ bb,
                                float* __restrict__ H,
                                float* __restrict__ Out,
                                int layer) {
    const int idx = blockIdx.x * 256 + threadIdx.x;
    const int n = idx & (NSZ - 1);
    const int c = (idx >> 12) & 127;
    const int b = idx >> 19;
    float xh = (T[idx] - mean[c]) * rsqrtf(var[c] + 1e-5f);
    float r = fmaxf(fmaf(xh, g[c], bb[c]), 0.0f);
    float hn = H[idx] + r;
    H[idx] = hn;
    Out[((size_t)b * (4 * CSZ) + (size_t)(layer * CSZ + c)) * NSZ + n] = hn;
}

// ---------------------------------------------------------------------------
// energy = qT' * qT (Gram), masked row softmax, output TRANSPOSED attention:
// attnT[b, m, n] (f16).  One block = one (b, 16-row n-block).  8 waves do the
// WMMA m-sweep into a 16x4096 f32 LDS strip (256 KB dynamic LDS, legal on
// CDNA5's 320 KB WGP LDS); per-row max/exp-sum with wave32 shuffles; then a
// transposed f16 write phase (32B per thread, absorbed by 192MB L2).
// ---------------------------------------------------------------------------
__global__ void energy_softmax_kernel(const _Float16* __restrict__ qT,
                                      const int* __restrict__ mask,
                                      _Float16* __restrict__ attnT) {
    extern __shared__ float strip[];   // 16 * NSZ floats
    __shared__ float invArr[16];
    const int b    = blockIdx.x >> 8;        // NSZ/16 == 256 blocks per batch
    const int n0   = (blockIdx.x & 255) * 16;
    const int lane = threadIdx.x & 31;
    const int wave = threadIdx.x >> 5;
    const int l15  = lane & 15;
    const int hi1  = lane >> 4;

    // A tile: rows n0..n0+15 of q^T, K = 32 channels (contiguous in qT)
    const int kbA = hi1 * 16;
    const _Float16* arow = qT + (size_t)(b * NSZ + n0 + l15) * 32;
    const v16h A = pack16(*(const v8h*)(arow + kbA),
                          *(const v8h*)(arow + 8 + kbA));

    int rmask[8];
#pragma unroll
    for (int v = 0; v < 8; ++v) rmask[v] = mask[b * NSZ + n0 + hi1 * 8 + v];

    const int kbB = hi1 * 16;
    for (int t = wave; t < NSZ / 16; t += 8) {
        const int m = t * 16 + l15;
        const _Float16* brow = qT + (size_t)(b * NSZ + m) * 32 + kbB;
        const v16h Bm = pack16(*(const v8h*)brow, *(const v8h*)(brow + 8));
        v8f c = {};
        c = __builtin_amdgcn_wmma_f32_16x16x32_f16(false, A, false, Bm,
                                                   (short)0, c, false, false);
        const int mm = mask[b * NSZ + m];
#pragma unroll
        for (int v = 0; v < 8; ++v) {
            const int row = hi1 * 8 + v;
            strip[row * NSZ + m] = (rmask[v] && mm) ? c[v] : NEGV;
        }
    }
    __syncthreads();

    // each wave finishes rows {wave, wave+8}: max, exp (in place), 1/sum
    for (int rr = 0; rr < 2; ++rr) {
        const int row = wave + rr * 8;
        float* er = strip + row * NSZ;
        float mx = -3.0e38f;
        for (int m = lane; m < NSZ; m += 32) mx = fmaxf(mx, er[m]);
#pragma unroll
        for (int off = 16; off > 0; off >>= 1) mx = fmaxf(mx, __shfl_xor(mx, off, 32));
        float s = 0.0f;
        for (int m = lane; m < NSZ; m += 32) {
            float ex = __expf(er[m] - mx);
            er[m] = ex;
            s += ex;
        }
#pragma unroll
        for (int off = 16; off > 0; off >>= 1) s += __shfl_xor(s, off, 32);
        if (lane == 0) invArr[row] = 1.0f / s;
    }
    __syncthreads();

    // transposed write: thread -> column m, 16 contiguous f16 (32B store)
    const size_t obase = (size_t)b * NSZ * NSZ + n0;
    for (int m = threadIdx.x; m < NSZ; m += 256) {
        v16h o;
#pragma unroll
        for (int v = 0; v < 16; ++v)
            o[v] = (_Float16)(strip[v * NSZ + m] * invArr[v]);
        *(v16h*)(attnT + obase + (size_t)m * NSZ) = o;
    }
}

// ---------------------------------------------------------------------------
// colsum[b,m] = sum_n attn[b,n,m] = row-sum of attnT  (contiguous, no atomics)
// One wave per m-row; 8 waves / block.
// ---------------------------------------------------------------------------
__global__ void colsum_kernel(const _Float16* __restrict__ attnT,
                              float* __restrict__ colsum) {
    const int wave = threadIdx.x >> 5;
    const int lane = threadIdx.x & 31;
    const int row = blockIdx.x * 8 + wave;        // flattened b*NSZ + m
    const _Float16* r = attnT + (size_t)row * NSZ;
    float s = 0.0f;
    for (int n = lane * 8; n < NSZ; n += 256) {
        v8h v = *(const v8h*)(r + n);
#pragma unroll
        for (int j = 0; j < 8; ++j) s += (float)v[j];
    }
#pragma unroll
    for (int off = 16; off > 0; off >>= 1) s += __shfl_xor(s, off, 32);
    if (lane == 0) colsum[row] = s;
}

// ---------------------------------------------------------------------------
// x_r GEMM:  x_r[b,c,m] = sum_n xv[b,c,n] * attnT[b,m,n]
// fused epilogue: d = h - x_r / (1e-9 + colsum[m])
// 8 waves/block, each wave -> 16x64 tile (4 f32 WMMA accumulators).
// All WMMA operands are contiguous 16B loads now (attnT).  K = 4096 / 32.
// grid = B * 8(cblk) * 8(m super-block of 512)
// ---------------------------------------------------------------------------
__global__ void xr_gemm_kernel(const _Float16* __restrict__ xv,
                               const _Float16* __restrict__ attnT,
                               const float* __restrict__ colsum,
                               const float* __restrict__ h,
                               float* __restrict__ dbuf) {
    int id = blockIdx.x;
    const int mblk = id & 7; id >>= 3;
    const int cblk = id & 7; id >>= 3;
    const int b = id;
    const int lane = threadIdx.x & 31;
    const int wave = threadIdx.x >> 5;
    const int l15 = lane & 15;
    const int hi1 = lane >> 4;
    const int c0 = cblk * 16;
    const int mbase = mblk * 512 + wave * 64;
    const int kbA = hi1 * 16;
    const int kbB = hi1 * 16;

    const _Float16* arowbase = xv + (size_t)(b * CSZ + c0 + l15) * NSZ;
    const _Float16* b0 = attnT + (size_t)b * NSZ * NSZ + (size_t)(mbase + l15) * NSZ + kbB;
    const _Float16* b1 = b0 + (size_t)16 * NSZ;
    const _Float16* b2 = b0 + (size_t)32 * NSZ;
    const _Float16* b3 = b0 + (size_t)48 * NSZ;

    v8f acc[4] = {};
#pragma unroll 2
    for (int k0 = 0; k0 < NSZ; k0 += 32) {
        const v16h A = pack16(*(const v8h*)(arowbase + k0 + kbA),
                              *(const v8h*)(arowbase + k0 + 8 + kbA));
        const v16h B0 = pack16(*(const v8h*)(b0 + k0), *(const v8h*)(b0 + k0 + 8));
        const v16h B1 = pack16(*(const v8h*)(b1 + k0), *(const v8h*)(b1 + k0 + 8));
        const v16h B2 = pack16(*(const v8h*)(b2 + k0), *(const v8h*)(b2 + k0 + 8));
        const v16h B3 = pack16(*(const v8h*)(b3 + k0), *(const v8h*)(b3 + k0 + 8));
        acc[0] = __builtin_amdgcn_wmma_f32_16x16x32_f16(false, A, false, B0, (short)0, acc[0], false, false);
        acc[1] = __builtin_amdgcn_wmma_f32_16x16x32_f16(false, A, false, B1, (short)0, acc[1], false, false);
        acc[2] = __builtin_amdgcn_wmma_f32_16x16x32_f16(false, A, false, B2, (short)0, acc[2], false, false);
        acc[3] = __builtin_amdgcn_wmma_f32_16x16x32_f16(false, A, false, B3, (short)0, acc[3], false, false);
    }

#pragma unroll
    for (int t = 0; t < 4; ++t) {
        const int m = mbase + t * 16 + l15;
        const float sc = 1.0f / (1e-9f + colsum[b * NSZ + m]);
#pragma unroll
        for (int v = 0; v < 8; ++v) {
            const int row = hi1 * 8 + v;
            const size_t idx = (size_t)(b * CSZ + c0 + row) * NSZ + m;
            dbuf[idx] = h[idx] - acc[t][v] * sc;
        }
    }
}

// ---------------------------------------------------------------------------
extern "C" void kernel_launch(void* const* d_in, const int* in_sizes, int n_in,
                              void* d_out, int out_size, void* d_ws, size_t ws_size,
                              hipStream_t stream) {
    (void)in_sizes; (void)n_in; (void)out_size; (void)ws_size;
    const float* x    = (const float*)d_in[0];
    const int*   mask = (const int*)d_in[1];
    const float* w1   = (const float*)d_in[2];
    const float* g1   = (const float*)d_in[3];
    const float* b1   = (const float*)d_in[4];
    const float* w2   = (const float*)d_in[5];
    const float* g2   = (const float*)d_in[6];
    const float* b2   = (const float*)d_in[7];
    const float* wqk  = (const float*)d_in[8];   // [4,32,128]
    const float* wv   = (const float*)d_in[9];   // [4,128,128]
    const float* bv   = (const float*)d_in[10];  // [4,128]
    const float* wt   = (const float*)d_in[11];  // [4,128,128]
    const float* bt   = (const float*)d_in[12];  // [4,128]
    const float* sg   = (const float*)d_in[13];  // [4,128]
    const float* sb   = (const float*)d_in[14];  // [4,128]
    float* out = (float*)d_out;

    const size_t BCN = (size_t)BSZ * CSZ * NSZ;      // 2M elems
    char* ws = (char*)d_ws;
    size_t off = 0;
    auto alloc = [&](size_t bytes) -> void* {
        void* p = ws + off;
        off += (bytes + 255) & ~(size_t)255;
        return p;
    };
    float*    h      = (float*)alloc(BCN * 4);                         // 8 MB
    float*    bufA   = (float*)alloc(BCN * 4);                         // 8 MB
    float*    bufB   = (float*)alloc(BCN * 4);                         // 8 MB
    _Float16* qT     = (_Float16*)alloc((size_t)BSZ * NSZ * 32 * 2);   // 1 MB
    _Float16* xv16   = (_Float16*)alloc(BCN * 2);                      // 4 MB
    float*    colsum = (float*)alloc((size_t)BSZ * NSZ * 4);           // 64 KB
    float*    meanb  = (float*)alloc(CSZ * 4);
    float*    varb   = (float*)alloc(CSZ * 4);
    _Float16* attnT  = (_Float16*)alloc((size_t)BSZ * NSZ * NSZ * 2);  // 128 MB

    const int elemBlocks = (int)(BCN / 256);          // 8192
    const int projBlocks = BSZ * 16 * 8;              // 512
    const size_t stripBytes = (size_t)16 * NSZ * 4;   // 256 KB dynamic LDS

    // ---- stage 1: two BN+ReLU projections ----
    proj128_kernel<<<projBlocks, 256, 0, stream>>>(w1, x, nullptr, bufA);
    bnstats_kernel<<<CSZ, 256, 0, stream>>>(bufA, meanb, varb);
    bnrelu_kernel<<<elemBlocks, 256, 0, stream>>>(bufA, meanb, varb, g1, b1, h);

    proj128_kernel<<<projBlocks, 256, 0, stream>>>(w2, h, nullptr, bufA);
    bnstats_kernel<<<CSZ, 256, 0, stream>>>(bufA, meanb, varb);
    bnrelu_kernel<<<elemBlocks, 256, 0, stream>>>(bufA, meanb, varb, g2, b2, h);

    // ---- 4 offset-attention layers (sequential) ----
    for (int i = 0; i < 4; ++i) {
        qproj_kernel<<<BSZ * 16 * 2, 256, 0, stream>>>(
            wqk + (size_t)i * 32 * CSZ, h, qT);
        proj128_f16_kernel<<<projBlocks, 256, 0, stream>>>(
            wv + (size_t)i * CSZ * CSZ, h, bv + (size_t)i * CSZ, xv16);

        energy_softmax_kernel<<<BSZ * (NSZ / 16), 256, stripBytes, stream>>>(
            qT, mask, attnT);

        colsum_kernel<<<(BSZ * NSZ) / 8, 256, 0, stream>>>(attnT, colsum);

        xr_gemm_kernel<<<BSZ * 8 * 8, 256, 0, stream>>>(xv16, attnT, colsum, h, bufA);

        proj128_kernel<<<projBlocks, 256, 0, stream>>>(
            wt + (size_t)i * CSZ * CSZ, bufA, bt + (size_t)i * CSZ, bufB);
        bnstats_kernel<<<CSZ, 256, 0, stream>>>(bufB, meanb, varb);
        residual_kernel<<<elemBlocks, 256, 0, stream>>>(
            bufB, meanb, varb, sg + (size_t)i * CSZ, sb + (size_t)i * CSZ,
            h, out, i);
    }
}
